// StructureAwareAttention_23897198035032
// MI455X (gfx1250) — matrix operand order repaired and checked by reading, not compiled, not run on hardware
//
#include <hip/hip_runtime.h>
#include <hip/hip_bf16.h>

// Structure-aware attention for MI455X (gfx1250, wave32, WMMA, async-to-LDS).
// Pipeline (all bf16 matmuls with f32 accumulation via v_wmma_f32_16x16x32_bf16):
//   1) qkv   = x  @ Wqkv + bqkv   -> Q,K [B,H,L,HD] bf16 ; V stored transposed [B,H,HD,L] bf16
//   2) SW    = sb @ Ws   + bs     -> [B*L, L] f32
//   3) attn  = softmax(QK^T*scale + SW) @ V  -> [B*L, D] bf16
//   4) out   = attn @ Wo + bo     -> d_out f32
// GEMMs double-buffer their LDS tiles with GLOBAL_LOAD_ASYNC_TO_LDS_B128 when
// the toolchain exposes the gfx1250 async builtins (s_wait_asynccnt sync).

typedef __bf16 bf16_t;
typedef __attribute__((ext_vector_type(16))) __bf16 v16bf;
typedef __attribute__((ext_vector_type(8)))  float  v8f;

#define B_  2
#define L_  1024
#define D_  1024
#define H_  16
#define HD_ 64

// ---- gfx1250 async copy availability (device pass only; host falls back) ----
#if defined(__AMDGCN__) && \
    __has_builtin(__builtin_amdgcn_global_load_async_to_lds_b128) && \
    __has_builtin(__builtin_amdgcn_s_wait_asynccnt)
#define USE_ASYNC_COPY 1
#else
#define USE_ASYNC_COPY 0
#endif

__device__ inline void async_cp16(void* lds_dst, const void* gsrc) {
#if USE_ASYNC_COPY
  typedef int v4i_ __attribute__((vector_size(4 * sizeof(int))));
  typedef __attribute__((address_space(1))) v4i_ as1_v4i;
  typedef __attribute__((address_space(3))) v4i_ as3_v4i;
  __builtin_amdgcn_global_load_async_to_lds_b128(
      (as1_v4i*)(unsigned long long)gsrc,
      (as3_v4i*)(unsigned int)(unsigned long long)lds_dst, 0, 0);
#else
  (void)lds_dst; (void)gsrc;
#endif
}

__device__ inline void wait_async0() {
#if USE_ASYNC_COPY
  __builtin_amdgcn_s_wait_asynccnt(0);
#endif
}

// ---- synchronous staging helpers (fallback path; widen to f32 in LDS) ------
__device__ inline void stage4(float* dst, const float* src) {
  float4 v = *(const float4*)src;
  dst[0] = v.x; dst[1] = v.y; dst[2] = v.z; dst[3] = v.w;
}
__device__ inline void stage4(float* dst, const bf16_t* src) {
  dst[0] = (float)src[0]; dst[1] = (float)src[1];
  dst[2] = (float)src[2]; dst[3] = (float)src[3];
}

// ---- generic tiled bf16 WMMA GEMM ------------------------------------------
// C[M,N] = A[M,K] @ Bw[K,N] + bias[N]
// MODE 0: write f32 row-major to Cout
// MODE 1: qkv epilogue: scatter to Q/K [B,H,L,HD] and V^T [B,H,HD,L] (bf16)
template<typename AT, int MODE>
__global__ __launch_bounds__(256)
void gemm_wmma_bf16(const AT* __restrict__ A, const float* __restrict__ Bw,
                    const float* __restrict__ bias, float* __restrict__ Cout,
                    bf16_t* __restrict__ Qo, bf16_t* __restrict__ Ko,
                    bf16_t* __restrict__ Vo, int M, int N, int K)
{
  constexpr int BM = 128, BN = 64, BK = 32;
  constexpr bool ASYNC_OK = (USE_ASYNC_COPY != 0) && (sizeof(AT) == 4);

  // f32 tiles, rows padded by 4 f32 (16B): async b128 chunks stay row-aligned,
  // A-fragment reads hit 16 distinct banks (m*36 mod 64 all distinct).
  __shared__ float Asf[2][BM][BK + 4];
  __shared__ float Bsf[2][BK][BN + 4];

  const int tid   = threadIdx.x;
  const int lane  = tid & 31, wave = tid >> 5;
  const int hi    = lane >> 4, ln = lane & 15;
  const int waveM = wave >> 1, waveN = wave & 1;
  const int rowBase = blockIdx.y * BM;
  const int colBase = blockIdx.x * BN;

  v8f acc[2][2] = {};

  const int arow = tid >> 3, acol = (tid & 7) * 4;   // 128x32 tile, 16B/thread/pass
  const int brow = tid >> 4, bcol = (tid & 15) * 4;  // 32x64 tile

  // async tile staging: copies go straight to LDS, no VGPR traffic
  auto stageAsync = [&](int k0, int buf) {
    #pragma unroll
    for (int rr = arow; rr < BM; rr += 32)
      async_cp16(&Asf[buf][rr][acol], &A[(size_t)(rowBase + rr) * K + k0 + acol]);
    #pragma unroll
    for (int rr = brow; rr < BK; rr += 16)
      async_cp16(&Bsf[buf][rr][bcol], &Bw[(size_t)(k0 + rr) * N + colBase + bcol]);
  };
  // synchronous fallback staging (also used for bf16 A, widening to f32)
  auto stageSync = [&](int k0, int buf) {
    #pragma unroll
    for (int rr = arow; rr < BM; rr += 32)
      stage4(&Asf[buf][rr][acol], &A[(size_t)(rowBase + rr) * K + k0 + acol]);
    #pragma unroll
    for (int rr = brow; rr < BK; rr += 16)
      stage4(&Bsf[buf][rr][bcol], &Bw[(size_t)(k0 + rr) * N + colBase + bcol]);
    if (k0 + BK < K) {  // global_prefetch_b8 for next tiles
      __builtin_prefetch(&A[(size_t)(rowBase + arow) * K + k0 + BK + acol], 0, 0);
      __builtin_prefetch(&Bw[(size_t)(k0 + BK + brow) * N + colBase + bcol], 0, 0);
    }
  };

  // one BK-step of WMMAs from LDS buffer `buf` (f32 -> bf16 at fragment read)
  auto computeTile = [&](int buf) {
    v16bf af[2], bfr[2];
    // A-fragments (16x32 bf16): K(j) = j + 16*hi
    #pragma unroll
    for (int fm = 0; fm < 2; ++fm) {
      const int m0 = waveM * 32 + fm * 16 + ln;
      #pragma unroll
      for (int j = 0; j < 16; ++j) {
        const int kk = j + hi * 16;
        af[fm][j] = (bf16_t)Asf[buf][m0][kk];
      }
    }
    // B-fragments (32x16 bf16): K(j) = j + 16*hi, N = ln
    #pragma unroll
    for (int fn = 0; fn < 2; ++fn) {
      const int c0 = waveN * 32 + fn * 16 + ln;
      #pragma unroll
      for (int j = 0; j < 16; ++j)
        bfr[fn][j] = (bf16_t)Bsf[buf][j + hi * 16][c0];
    }
    #pragma unroll
    for (int fm = 0; fm < 2; ++fm)
      #pragma unroll
      for (int fn = 0; fn < 2; ++fn)
        acc[fm][fn] = __builtin_amdgcn_wmma_f32_16x16x32_bf16(
            false, af[fm], false, bfr[fn], (short)0, acc[fm][fn], false, false);
  };

  if constexpr (ASYNC_OK) {
    // double-buffered: async copy of tile i+1 overlaps WMMAs on tile i
    const int T = K / BK;
    stageAsync(0, 0);
    wait_async0();
    __syncthreads();
    for (int it = 0; it < T; ++it) {
      const int cur = it & 1;
      if (it + 1 < T) stageAsync((it + 1) * BK, cur ^ 1);
      computeTile(cur);
      wait_async0();      // next tile landed in LDS
      __syncthreads();    // all waves' reads of `cur` done; copies visible
    }
  } else {
    for (int k0 = 0; k0 < K; k0 += BK) {
      stageSync(k0, 0);
      __syncthreads();
      computeTile(0);
      __syncthreads();
    }
  }

  // epilogue: C/D layout -> row = g + 8*hi, col = ln
  #pragma unroll
  for (int fm = 0; fm < 2; ++fm) {
    #pragma unroll
    for (int fn = 0; fn < 2; ++fn) {
      const int c  = colBase + waveN * 32 + fn * 16 + ln;
      const float bv = bias ? bias[c] : 0.0f;
      #pragma unroll
      for (int g = 0; g < 8; ++g) {
        const int r = rowBase + waveM * 32 + fm * 16 + g + 8 * hi;
        const float val = acc[fm][fn][g] + bv;
        if constexpr (MODE == 0) {
          Cout[(size_t)r * N + c] = val;
        } else {
          // col decomposition of 3*D feature: (which, head, dim)
          const int w3 = c >> 10, h = (c >> 6) & (H_ - 1), d = c & (HD_ - 1);
          const int b = r >> 10, l = r & (L_ - 1);
          const bf16_t bvv = (bf16_t)val;
          if (w3 == 0)      Qo[(((size_t)(b * H_ + h)) * L_ + l) * HD_ + d] = bvv;
          else if (w3 == 1) Ko[(((size_t)(b * H_ + h)) * L_ + l) * HD_ + d] = bvv;
          else              Vo[(((size_t)(b * H_ + h)) * HD_ + d) * L_ + l] = bvv; // V^T
        }
      }
    }
  }
}

// ---- fused attention: softmax(Q K^T * scale + SW) @ V -----------------------
// Block = 256 threads (8 waves) handles one (b, h, 16-query-row tile).
// Wave w owns keys [w*128, w*128+128): scores live in registers (64 f32/lane),
// cross-lane row stats via shuffles, cross-wave via small LDS, P transposed to
// A-fragment layout through a padded per-wave LDS tile, O reduced through LDS.
__global__ __launch_bounds__(256)
void attn_wmma(const bf16_t* __restrict__ Q, const bf16_t* __restrict__ Kt,
               const bf16_t* __restrict__ Vt, const float* __restrict__ SW,
               bf16_t* __restrict__ Aout)
{
  __shared__ float  PO[8][16][HD_];     // 32 KB partial O
  __shared__ bf16_t Pst[8][16][34];     // per-wave P transpose staging (+2 pad)
  __shared__ float  redMax[16][8];
  __shared__ float  redSum[16][8];

  const int tid  = threadIdx.x;
  const int lane = tid & 31, wave = tid >> 5;
  const int hi   = lane >> 4, ln = lane & 15;
  const int q0   = blockIdx.x * 16;
  const int h    = blockIdx.y, b = blockIdx.z;
  const int kb0  = wave * 128;

  const size_t headOff = ((size_t)(b * H_ + h)) * L_ * HD_;

  // Q A-fragments for K-dim (head dim) 0..31 and 32..63
  v16bf qa[2];
  #pragma unroll
  for (int f = 0; f < 2; ++f)
    #pragma unroll
    for (int j = 0; j < 16; ++j) {
      const int kk = f * 32 + hi * 16 + j;
      qa[f][j] = Q[headOff + (size_t)(q0 + ln) * HD_ + kk];
    }

  // ---- scores: S = Q K^T * scale + SW, kept in registers --------------------
  const float scale = 0.125f;  // 1/sqrt(64)
  float sreg[8][8];
  #pragma unroll
  for (int f = 0; f < 8; ++f) {
    const int kb = kb0 + f * 16;
    v16bf kv0, kv1;  // B-frag of K^T: N = key, K-dim = head dim (contiguous load)
    #pragma unroll
    for (int j = 0; j < 16; ++j) {
      const size_t kr = headOff + (size_t)(kb + ln) * HD_;
      kv0[j] = Kt[kr + hi * 16 + j];
      kv1[j] = Kt[kr + 32 + hi * 16 + j];
    }
    v8f s = {};
    s = __builtin_amdgcn_wmma_f32_16x16x32_bf16(false, qa[0], false, kv0, (short)0, s, false, false);
    s = __builtin_amdgcn_wmma_f32_16x16x32_bf16(false, qa[1], false, kv1, (short)0, s, false, false);
    #pragma unroll
    for (int g = 0; g < 8; ++g) {
      const int qrow = q0 + g + 8 * hi;
      sreg[f][g] = s[g] * scale + SW[((size_t)(b * L_ + qrow)) * L_ + kb + ln];
    }
  }

  // ---- row max (wave-local 128 keys, then cross-wave) -----------------------
  float pm[8];
  #pragma unroll
  for (int g = 0; g < 8; ++g) {
    float m = sreg[0][g];
    #pragma unroll
    for (int f = 1; f < 8; ++f) m = fmaxf(m, sreg[f][g]);
    #pragma unroll
    for (int off = 8; off > 0; off >>= 1) m = fmaxf(m, __shfl_xor(m, off, 32));
    pm[g] = m;
  }
  if (ln == 0) {
    #pragma unroll
    for (int g = 0; g < 8; ++g) redMax[g + 8 * hi][wave] = pm[g];
  }
  __syncthreads();
  float rowM[8];
  #pragma unroll
  for (int g = 0; g < 8; ++g) {
    float m = redMax[g + 8 * hi][0];
    #pragma unroll
    for (int ww = 1; ww < 8; ++ww) m = fmaxf(m, redMax[g + 8 * hi][ww]);
    rowM[g] = m;
  }

  // ---- exponentiate + row sums ----------------------------------------------
  #pragma unroll
  for (int g = 0; g < 8; ++g) {
    float sum = 0.f;
    #pragma unroll
    for (int f = 0; f < 8; ++f) {
      sreg[f][g] = __expf(sreg[f][g] - rowM[g]);
      sum += sreg[f][g];
    }
    #pragma unroll
    for (int off = 8; off > 0; off >>= 1) sum += __shfl_xor(sum, off, 32);
    if (ln == 0) redSum[g + 8 * hi][wave] = sum;
  }

  // ---- P @ V over this wave's 128 keys (4 chunks of K=32) -------------------
  v8f oacc[4] = {};
  #pragma unroll
  for (int c = 0; c < 4; ++c) {
    #pragma unroll
    for (int part = 0; part < 2; ++part) {
      const int f = c * 2 + part;
      #pragma unroll
      for (int g = 0; g < 8; ++g)
        Pst[wave][g + 8 * hi][part * 16 + ln] = (bf16_t)sreg[f][g];
    }
    __syncthreads();                       // order LDS store -> cross-lane load
    v16bf pa;                              // P as A-fragment (16 rows x 32 keys)
    #pragma unroll
    for (int j = 0; j < 16; ++j) {
      const int kk = hi * 16 + j;
      pa[j] = Pst[wave][ln][kk];
    }
    #pragma unroll
    for (int nf = 0; nf < 4; ++nf) {       // V^T gives contiguous B-frag loads
      v16bf vv;
      const int d = nf * 16 + ln;
      #pragma unroll
      for (int j = 0; j < 16; ++j) {
        const int key = kb0 + c * 32 + hi * 16 + j;
        vv[j] = Vt[headOff + (size_t)d * L_ + key];
      }
      oacc[nf] = __builtin_amdgcn_wmma_f32_16x16x32_bf16(
          false, pa, false, vv, (short)0, oacc[nf], false, false);
    }
    __syncthreads();
  }

  // ---- reduce partial O across waves, normalize, store ----------------------
  #pragma unroll
  for (int nf = 0; nf < 4; ++nf)
    #pragma unroll
    for (int g = 0; g < 8; ++g)
      PO[wave][g + 8 * hi][nf * 16 + ln] = oacc[nf][g];
  __syncthreads();

  for (int idx = tid; idx < 16 * HD_; idx += 256) {
    const int m = idx >> 6, d = idx & (HD_ - 1);
    float acc = 0.f, rs = 0.f;
    #pragma unroll
    for (int ww = 0; ww < 8; ++ww) { acc += PO[ww][m][d]; rs += redSum[m][ww]; }
    Aout[((size_t)(b * L_ + q0 + m)) * D_ + h * HD_ + d] = (bf16_t)(acc / rs);
  }
}

// ---- host launcher ----------------------------------------------------------
extern "C" void kernel_launch(void* const* d_in, const int* in_sizes, int n_in,
                              void* d_out, int out_size, void* d_ws, size_t ws_size,
                              hipStream_t stream) {
  const float* x    = (const float*)d_in[0];
  const float* sb   = (const float*)d_in[1];
  const float* Wqkv = (const float*)d_in[2];
  const float* bqkv = (const float*)d_in[3];
  const float* Ws   = (const float*)d_in[4];
  const float* bsv  = (const float*)d_in[5];
  const float* Wo   = (const float*)d_in[6];
  const float* bo   = (const float*)d_in[7];
  float* out = (float*)d_out;

  // workspace layout: Q(4MB) K(4MB) V^T(4MB) SW(8MB) attnOut(4MB) = 24 MB
  char* w = (char*)d_ws;
  const size_t headBytes = (size_t)B_ * H_ * L_ * HD_ * sizeof(bf16_t);
  bf16_t* Qb = (bf16_t*)w; w += headBytes;
  bf16_t* Kb = (bf16_t*)w; w += headBytes;
  bf16_t* Vb = (bf16_t*)w; w += headBytes;
  float*  SW = (float*)w;  w += (size_t)B_ * L_ * L_ * sizeof(float);
  bf16_t* AO = (bf16_t*)w; w += (size_t)B_ * L_ * D_ * sizeof(bf16_t);

  const int M = B_ * L_;  // 2048

  // 1) QKV projection + scatter into attention-friendly layouts
  gemm_wmma_bf16<float, 1><<<dim3(3 * D_ / 64, M / 128), 256, 0, stream>>>(
      x, Wqkv, bqkv, nullptr, Qb, Kb, Vb, M, 3 * D_, D_);

  // 2) structure-bias projection -> SW [B*L, L] f32
  gemm_wmma_bf16<float, 0><<<dim3(L_ / 64, M / 128), 256, 0, stream>>>(
      sb, Ws, bsv, SW, nullptr, nullptr, nullptr, M, L_, D_);

  // 3) fused biased softmax attention
  attn_wmma<<<dim3(L_ / 16, H_, B_), 256, 0, stream>>>(Qb, Kb, Vb, SW, AO);

  // 4) output projection -> d_out f32
  gemm_wmma_bf16<bf16_t, 0><<<dim3(D_ / 64, M / 128), 256, 0, stream>>>(
      AO, Wo, bo, out, nullptr, nullptr, nullptr, M, D_, D_);
}
